// Attention_22256520528497
// MI455X (gfx1250) — matrix operand order repaired and checked by reading, not compiled, not run on hardware
//
#include <hip/hip_runtime.h>

// ---------------- types ----------------
typedef __attribute__((ext_vector_type(8)))  __bf16 v8bf;
typedef __attribute__((ext_vector_type(16))) __bf16 v16bf;
typedef __attribute__((ext_vector_type(8)))  float  v8f;
typedef __attribute__((ext_vector_type(4)))  int    v4i;
typedef unsigned long long u64;

typedef __attribute__((address_space(1))) v4i as1_v4i;  // global int4
typedef __attribute__((address_space(3))) v4i as3_v4i;  // LDS int4

static constexpr int Bb = 2, Ss = 2048, Dd = 1024, Hh = 16, HDim = 64;
static constexpr int Mtot = Bb * Ss;   // 4096

#if __has_builtin(__builtin_amdgcn_global_load_async_to_lds_b128)
#define HAVE_ASYNC_LDS 1
#else
#define HAVE_ASYNC_LDS 0
#endif

// 16-byte global -> LDS copy (async on CDNA5: GLOBAL_LOAD_ASYNC_TO_LDS_B128)
__device__ inline void cp16_g2l(void* lds, const void* g) {
#if HAVE_ASYNC_LDS
  __builtin_amdgcn_global_load_async_to_lds_b128(
      (as1_v4i*)(u64)g, (as3_v4i*)(unsigned)(u64)lds, 0, 0);
#else
  ((u64*)lds)[0] = ((const u64*)g)[0];
  ((u64*)lds)[1] = ((const u64*)g)[1];
#endif
}
__device__ inline void async_join() {
#if HAVE_ASYNC_LDS
#if __has_builtin(__builtin_amdgcn_s_wait_asynccnt)
  __builtin_amdgcn_s_wait_asynccnt(0);
#else
  asm volatile("s_wait_asynccnt 0x0" ::: "memory");
#endif
#endif
}

// ---------------- WMMA fragment helpers ----------------
__device__ inline v16bf cat8(v8bf lo, v8bf hi) {
  return __builtin_shufflevector(lo, hi, 0,1,2,3,4,5,6,7,8,9,10,11,12,13,14,15);
}

// A-operand (M x K), loaded from row-major [row][k] LDS tile.
// lane: M = lane&15 ; element e=8h+j -> K = kb + 16h + 8g + j, g = lane>>4
__device__ inline v16bf load_a(const __bf16* base, int row0, int ld, int kb, int lane) {
  const int g = lane >> 4, r = lane & 15;
  const __bf16* p = base + (row0 + r) * ld + kb + 8 * g;
  v8bf lo = *(const v8bf*)p;        // K = kb + 8g + j
  v8bf hi = *(const v8bf*)(p + 16); // K = kb + 16 + 8g + j
  return cat8(lo, hi);
}

// B-operand (K x N). We load from its transpose BT=[n][k] row-major in LDS.
// lane: N = lane&15 ; element e=8h+j -> K = kb + 16h + 8g + j, g = lane>>4
__device__ inline v16bf load_b(const __bf16* base, int row0, int ld, int kb, int lane) {
  const int g = lane >> 4, n = lane & 15;
  const __bf16* p = base + (row0 + n) * ld + kb + 8 * g;
  v8bf lo = *(const v8bf*)p;
  v8bf hi = *(const v8bf*)(p + 16);
  return cat8(lo, hi);
}

__device__ inline v8f wmma_bf16(v16bf a, v16bf b, v8f c) {
  return __builtin_amdgcn_wmma_f32_16x16x32_bf16(false, a, false, b, (short)0, c, false, false);
}

// ---------------- staging / store helpers ----------------
__device__ inline void stage4(__bf16* dst, const float* src) {
  const float4 v = *(const float4*)src;
  dst[0] = (__bf16)v.x; dst[1] = (__bf16)v.y;
  dst[2] = (__bf16)v.z; dst[3] = (__bf16)v.w;
}
__device__ inline void stage4(__bf16* dst, const __bf16* src) {
  *(u64*)dst = *(const u64*)src;
}
__device__ inline void store_c(float* p, float v)  { *p = v; }
__device__ inline void store_c(__bf16* p, float v) { *p = (__bf16)v; }

// ---------------- GEMM: C[M,N] = A[M,K] * W[N,K]^T + bias[N] ----------------
// block = 128 threads (4 waves), workgroup tile 128(M) x 64(N), K chunk 32.
// Each wave owns 32 M-rows: 2 A-frags x 4 B-frags = 8 WMMAs per chunk.
template <typename AT, typename CT>
__global__ __launch_bounds__(128)
void gemm_bias(const AT* __restrict__ A, const float* __restrict__ W,
               const float* __restrict__ bias, CT* __restrict__ C,
               int M, int N, int K) {
  __shared__ __bf16 As[128 * 32];
  __shared__ __bf16 Ws[64 * 32];
  const int tid  = threadIdx.x;
  const int lane = tid & 31;
  const int wave = tid >> 5;
  const int m0 = blockIdx.y << 7;
  const int n0 = blockIdx.x << 6;

  v8f acc[2][4];
#pragma unroll
  for (int p = 0; p < 2; ++p)
#pragma unroll
    for (int t = 0; t < 4; ++t)
#pragma unroll
      for (int i = 0; i < 8; ++i) acc[p][t][i] = 0.f;

  for (int k0 = 0; k0 < K; k0 += 32) {
    // stage 128x32 A tile and 64x32 W tile (bf16) into LDS
#pragma unroll
    for (int i = tid; i < 1024; i += 128) {
      const int row = i >> 3, c4 = (i & 7) << 2;
      stage4(&As[row * 32 + c4], A + (size_t)(m0 + row) * K + k0 + c4);
    }
#pragma unroll
    for (int i = tid; i < 512; i += 128) {
      const int row = i >> 3, c4 = (i & 7) << 2;
      stage4(&Ws[row * 32 + c4], W + (size_t)(n0 + row) * K + k0 + c4);
    }
    __syncthreads();
    const v16bf a0 = load_a(As, (wave << 5),      32, 0, lane);
    const v16bf a1 = load_a(As, (wave << 5) + 16, 32, 0, lane);
#pragma unroll
    for (int t = 0; t < 4; ++t) {
      const v16bf bf = load_b(Ws, t << 4, 32, 0, lane);
      acc[0][t] = wmma_bf16(a0, bf, acc[0][t]);
      acc[1][t] = wmma_bf16(a1, bf, acc[1][t]);
    }
    __syncthreads();
  }

  // epilogue: C lane layout = col(N)=lane&15, row(M)=i+8*(lane>>4)
  const int r = lane & 15, g = lane >> 4;
#pragma unroll
  for (int p = 0; p < 2; ++p)
#pragma unroll
    for (int t = 0; t < 4; ++t) {
      const float bv = bias[n0 + (t << 4) + r];
#pragma unroll
      for (int i = 0; i < 8; ++i) {
        const int mm = m0 + (wave << 5) + (p << 4) + i + 8 * g;
        store_c(&C[(size_t)mm * N + n0 + (t << 4) + r], acc[p][t][i] + bv);
      }
    }
}

// ---------------- RoPE + head packing ----------------
// qkv bf16 [M][3D]  ->  qh [bh][s][hd] (scaled by 1/8), kh [bh][s][hd], vt [bh][hd][s]
__global__ __launch_bounds__(256)
void rope_pack(const __bf16* __restrict__ qkv, const float* __restrict__ rot,
               __bf16* __restrict__ qh, __bf16* __restrict__ kh,
               __bf16* __restrict__ vt) {
  const int idx = blockIdx.x * 256 + threadIdx.x; // B*S*H*32 = 2,097,152 exact
  const int j = idx & 31;
  const int h = (idx >> 5) & 15;
  const int s = (idx >> 9) & 2047;
  const int b = idx >> 20;
  const size_t row = (size_t)(b * Ss + s) * (3 * Dd);
  const int col = h * HDim + 2 * j;
  const float qe = (float)qkv[row + col],        qo = (float)qkv[row + col + 1];
  const float ke = (float)qkv[row + Dd + col],   ko = (float)qkv[row + Dd + col + 1];
  const float ve = (float)qkv[row + 2*Dd + col], vo = (float)qkv[row + 2*Dd + col + 1];
  const float c  = rot[(s * 32 + j) * 2];
  const float sn = rot[(s * 32 + j) * 2 + 1];
  const int bh = b * Hh + h;
  const size_t qrow = ((size_t)bh * Ss + s) * HDim + 2 * j;
  qh[qrow]     = (__bf16)(0.125f * (qe * c - qo * sn));
  qh[qrow + 1] = (__bf16)(0.125f * (qe * sn + qo * c));
  kh[qrow]     = (__bf16)(ke * c - ko * sn);
  kh[qrow + 1] = (__bf16)(ke * sn + ko * c);
  const size_t vrow = ((size_t)bh * HDim + 2 * j) * Ss + s;
  vt[vrow]      = (__bf16)ve;
  vt[vrow + Ss] = (__bf16)vo;
}

// ---------------- Flash attention ----------------
// grid: (S/64, B*H); block 128 (4 waves, 16 queries each).
// S^T = K @ Q^T  (queries in lanes) ; O^T = V^T @ P^T (accumulated in regs)
__global__ __launch_bounds__(128)
void flash_attn(const __bf16* __restrict__ qh, const __bf16* __restrict__ kh,
                const __bf16* __restrict__ vt, __bf16* __restrict__ ctx) {
  const int bh = blockIdx.y;
  const int b  = bh >> 4;
  const int h  = bh & 15;
  const int q0 = blockIdx.x << 6;
  const int tid = threadIdx.x, lane = tid & 31, wave = tid >> 5;
  const int g = lane >> 4, r = lane & 15;

  __shared__ __bf16 Qs[64 * 64];        // [q][hd]
  __shared__ __bf16 Ks[64 * 64];        // [key][hd]
  __shared__ __bf16 Vs[64 * 64];        // V^T chunk: [hd][key]
  __shared__ __bf16 Ps[4 * 16 * 64];    // per-wave P^T staging: [wave][q][key]

  // load Q tile (already scaled by 1/sqrt(HD)) via async copy
  const size_t qbase = ((size_t)bh * Ss + q0) * HDim;
#pragma unroll
  for (int i = tid; i < 512; i += 128)            // 512 x 16B granules
    cp16_g2l(&Qs[i * 8], qh + qbase + i * 8);
  async_join();
  __syncthreads();
  const v16bf bq0 = load_b(Qs, wave << 4, 64, 0,  lane);
  const v16bf bq1 = load_b(Qs, wave << 4, 64, 32, lane);

  v8f o[4];
#pragma unroll
  for (int d = 0; d < 4; ++d)
#pragma unroll
    for (int i = 0; i < 8; ++i) o[d][i] = 0.f;
  float m_prev = -3.0e38f;
  float l = 0.f;

  for (int kc = 0; kc < Ss / 64; ++kc) {
    __syncthreads();
    const size_t kbase = ((size_t)bh * Ss + (size_t)kc * 64) * HDim;
    const size_t vbase = (size_t)bh * HDim * Ss + (size_t)kc * 64;
#pragma unroll
    for (int i = tid; i < 512; i += 128)
      cp16_g2l(&Ks[i * 8], kh + kbase + i * 8);
#pragma unroll
    for (int i = tid; i < 512; i += 128) {
      const int hd = i >> 3, seg = (i & 7) << 3;
      cp16_g2l(&Vs[i * 8], vt + vbase + (size_t)hd * Ss + seg);
    }
    // prefetch next chunk into L2 while async engine fills LDS
    if (kc + 1 < Ss / 64) {
      __builtin_prefetch(kh + kbase + 4096 + tid * 32, 0, 1);
      __builtin_prefetch(vt + vbase + 64 + (size_t)(tid & 63) * Ss, 0, 1);
    }
    async_join();
    __syncthreads();

    // scores: S^T[key, q] for 64 keys x 16 queries, HD=64 -> 2 wmmas per tile
    v8f s[4];
#pragma unroll
    for (int t = 0; t < 4; ++t) {
      v8f z;
#pragma unroll
      for (int i = 0; i < 8; ++i) z[i] = 0.f;
      z    = wmma_bf16(load_a(Ks, t << 4, 64, 0,  lane), bq0, z);
      s[t] = wmma_bf16(load_a(Ks, t << 4, 64, 32, lane), bq1, z);
    }

    // online softmax over keys (per query = per lane column)
    float mc = -3.0e38f;
#pragma unroll
    for (int t = 0; t < 4; ++t)
#pragma unroll
      for (int i = 0; i < 8; ++i) mc = fmaxf(mc, s[t][i]);
    mc = fmaxf(mc, __shfl_xor(mc, 16));
    const float m_new = fmaxf(m_prev, mc);
    const float alpha = __expf(m_prev - m_new);
    float rs = 0.f;
#pragma unroll
    for (int t = 0; t < 4; ++t)
#pragma unroll
      for (int i = 0; i < 8; ++i) {
        const float p = __expf(s[t][i] - m_new);
        s[t][i] = p;
        rs += p;
      }
    rs += __shfl_xor(rs, 16);
    l = l * alpha + rs;
    m_prev = m_new;
#pragma unroll
    for (int d = 0; d < 4; ++d)
#pragma unroll
      for (int i = 0; i < 8; ++i) o[d][i] *= alpha;

    // transpose P through per-wave LDS into B-operand layout [q][key]
#pragma unroll
    for (int t = 0; t < 4; ++t)
#pragma unroll
      for (int i = 0; i < 8; ++i)
        Ps[wave * 1024 + r * 64 + (t << 4) + 8 * g + i] = (__bf16)s[t][i];

    const v16bf bp0 = load_b(Ps + wave * 1024, 0, 64, 0,  lane);
    const v16bf bp1 = load_b(Ps + wave * 1024, 0, 64, 32, lane);

    // O^T += V^T @ P^T : 4 hd sub-tiles, 64-key chunk = 2 wmmas each
#pragma unroll
    for (int d = 0; d < 4; ++d) {
      o[d] = wmma_bf16(load_a(Vs, d << 4, 64, 0,  lane), bp0, o[d]);
      o[d] = wmma_bf16(load_a(Vs, d << 4, 64, 32, lane), bp1, o[d]);
    }
  }

  // finalize: O^T lane layout = col(q)=lane&15, row(hd)=i+8g per sub-tile
  const float inv = 1.0f / l;
  const size_t row = (size_t)b * Ss + q0 + (wave << 4) + r;
#pragma unroll
  for (int d = 0; d < 4; ++d)
#pragma unroll
    for (int i = 0; i < 8; ++i) {
      const int hd = (d << 4) + i + 8 * g;
      ctx[row * Dd + h * HDim + hd] = (__bf16)(o[d][i] * inv);
    }
}

// ---------------- launch ----------------
extern "C" void kernel_launch(void* const* d_in, const int* in_sizes, int n_in,
                              void* d_out, int out_size, void* d_ws, size_t ws_size,
                              hipStream_t stream) {
  const float* x     = (const float*)d_in[0];
  const float* rot   = (const float*)d_in[1];
  const float* w_qkv = (const float*)d_in[2];
  const float* b_qkv = (const float*)d_in[3];
  const float* w_o   = (const float*)d_in[4];
  const float* b_o   = (const float*)d_in[5];
  float* out = (float*)d_out;

  char* ws = (char*)d_ws;
  __bf16* qkvb = (__bf16*)(ws);                       // 4096*3072 bf16 = 24 MB
  __bf16* qh   = (__bf16*)(ws + 25165824);            // 8 MB
  __bf16* kh   = (__bf16*)(ws + 33554432);            // 8 MB
  __bf16* vt   = (__bf16*)(ws + 41943040);            // 8 MB
  __bf16* ctx  = (__bf16*)(ws + 50331648);            // 8 MB

  // 1) QKV projection
  gemm_bias<float, __bf16><<<dim3(3 * Dd / 64, Mtot / 128), 128, 0, stream>>>(
      x, w_qkv, b_qkv, qkvb, Mtot, 3 * Dd, Dd);
  // 2) RoPE + head packing (+1/sqrt(HD) folded into Q)
  rope_pack<<<(Bb * Ss * Hh * 32) / 256, 256, 0, stream>>>(qkvb, rot, qh, kh, vt);
  // 3) Flash attention
  flash_attn<<<dim3(Ss / 64, Bb * Hh), 128, 0, stream>>>(qh, kh, vt, ctx);
  // 4) Output projection
  gemm_bias<__bf16, float><<<dim3(Dd / 64, Mtot / 128), 128, 0, stream>>>(
      ctx, w_o, b_o, out, Mtot, Dd, Dd);
}
